// DeformableTransformerDecoderLayer_83958020702436
// MI455X (gfx1250) — matrix operand order repaired and checked by reading, not compiled, not run on hardware
//
#include <hip/hip_runtime.h>
#include <hip/hip_bf16.h>

// ---------------- problem constants ----------------
#define DD   256
#define FF   1024
#define NH   8
#define HD   32
#define NLV  4
#define NP   4
#define BSZ  4
#define NQ   300
#define NKK  17
#define LQ   (NQ * NKK)          // 5100
#define NTOK (BSZ * NQ * NKK)    // 20400
#define LV   11253
#define MVAL (BSZ * LV)          // 45012

typedef __attribute__((ext_vector_type(16))) _Float16 v16h;
typedef __attribute__((ext_vector_type(8)))  float    v8f;

// =====================================================================
// Generic GEMM: C[M,N] = act( A[M,K] @ W[N,K]^T + bias + (Cin) )
// One wave per 16x64 tile: 4 accumulators share one A fragment ->
// 4 independent v_wmma_f32_16x16x32_f16 chains per K-step.
// K is a template constant (256 or 1024) so the K loop fully unrolls.
// act: 0=none, 1=relu, 2=sigmoid
// =====================================================================
template<int K>
__launch_bounds__(256)
__global__ void gemm_wmma_kernel(const float* __restrict__ A, int lda,
                                 const float* __restrict__ W, int ldw,
                                 const float* __restrict__ bias,
                                 const float* __restrict__ Cin,
                                 float* __restrict__ C, int ldc,
                                 int M, int N, int act) {
    int wave = blockIdx.x * 8 + (threadIdx.x >> 5);
    int lane = threadIdx.x & 31;
    int mtiles = (M + 15) >> 4;
    int ntiles = N >> 6;                         // 64-wide N tiles
    if (wave >= mtiles * ntiles) return;
    int tm = (wave % mtiles) << 4;
    int tn = (wave / mtiles) << 6;

    int l15 = lane & 15;
    int lhi = lane >> 4;                         // 0 or 1
    int arow = tm + l15; if (arow >= M) arow = M - 1;   // clamp tail rows
    // A 16-bit 16x32 layout: lane<16 -> K {0..7,16..23}; lane>=16 -> K {8..15,24..31}
    const float* Ap = A + (size_t)arow * lda + lhi * 8;
    // B 16-bit 32x16 layout: lane<16 -> K {0..7,16..23}; lane>=16 -> K {8..15,24..31}
    const float* Wp[4];
#pragma unroll
    for (int j = 0; j < 4; ++j)
        Wp[j] = W + (size_t)(tn + j * 16 + l15) * ldw + lhi * 8;

    v8f cc[4];
#pragma unroll
    for (int j = 0; j < 4; ++j)
#pragma unroll
        for (int r = 0; r < 8; ++r) cc[j][r] = 0.0f;

#pragma unroll
    for (int k0 = 0; k0 < K; k0 += 32) {
        __builtin_prefetch(Ap + k0 + 64, 0, 1);
        v16h afr;
#pragma unroll
        for (int v = 0; v < 8; ++v) {
            int g = v >> 2, vv = v & 3;          // K = k0 + g*16 + lhi*8 + vv*2 (+0/1)
            float2 f = *(const float2*)(Ap + k0 + g * 16 + vv * 2);
            afr[2 * v]     = (_Float16)f.x;
            afr[2 * v + 1] = (_Float16)f.y;
        }
#pragma unroll
        for (int j = 0; j < 4; ++j) {
            v16h bfr;
#pragma unroll
            for (int v = 0; v < 4; ++v) {
                float4 f = *(const float4*)(Wp[j] + k0 + (v >> 1) * 16 + (v & 1) * 4);
                bfr[4 * v]     = (_Float16)f.x;
                bfr[4 * v + 1] = (_Float16)f.y;
                bfr[4 * v + 2] = (_Float16)f.z;
                bfr[4 * v + 3] = (_Float16)f.w;
            }
            cc[j] = __builtin_amdgcn_wmma_f32_16x16x32_f16(false, afr, false, bfr,
                                                           (short)0, cc[j], false, false);
        }
    }

    int mb = tm + lhi * 8;
#pragma unroll
    for (int j = 0; j < 4; ++j) {
        int n  = tn + j * 16 + l15;
        float bn = bias ? bias[n] : 0.0f;
#pragma unroll
        for (int r = 0; r < 8; ++r) {
            int m = mb + r;
            if (m < M) {
                float v = cc[j][r] + bn;
                if (Cin) v += Cin[(size_t)m * ldc + n];
                if (act == 1)      v = fmaxf(v, 0.0f);
                else if (act == 2) v = 1.0f / (1.0f + __expf(-v));
                C[(size_t)m * ldc + n] = v;
            }
        }
    }
}

// =====================================================================
// elementwise add: out = a + b
// =====================================================================
__global__ void add_kernel(const float* __restrict__ a, const float* __restrict__ b,
                           float* __restrict__ out, int n) {
    int i = blockIdx.x * blockDim.x + threadIdx.x;
    if (i < n) out[i] = a[i] + b[i];
}

// =====================================================================
// Within-attention: sequences of length NKK=17 (one block per (b,nq) group)
// qkv layout: [token][768] = [q(256) | k(256) | v(256)], token = grp*17 + kk
// =====================================================================
__launch_bounds__(256)
__global__ void attn_within_kernel(const float* __restrict__ qkv, float* __restrict__ out) {
    __shared__ float sQ[NKK * DD], sK[NKK * DD], sV[NKK * DD];
    int grp = blockIdx.x;                               // 0 .. BSZ*NQ-1
    size_t base = (size_t)grp * NKK * (3 * DD);
    for (int i = threadIdx.x; i < NKK * DD; i += blockDim.x) {
        int tok = i / DD, d = i % DD;
        const float* row = qkv + base + (size_t)tok * (3 * DD);
        sQ[i] = row[d];
        sK[i] = row[DD + d];
        sV[i] = row[2 * DD + d];
    }
    __syncthreads();
    int unit = threadIdx.x;                             // h*NKK + qi ; 136 active
    if (unit < NH * NKK) {
        int h = unit / NKK, qi = unit % NKK;
        const float scale = 0.17677669529663687f;       // 1/sqrt(32)
        float q[HD];
#pragma unroll
        for (int d = 0; d < HD; ++d) q[d] = sQ[qi * DD + h * HD + d];
        float s[NKK];
        float m = -1e30f;
#pragma unroll
        for (int kk = 0; kk < NKK; ++kk) {
            float acc = 0.0f;
#pragma unroll
            for (int d = 0; d < HD; ++d) acc += q[d] * sK[kk * DD + h * HD + d];
            s[kk] = acc * scale;
            m = fmaxf(m, s[kk]);
        }
        float l = 0.0f;
#pragma unroll
        for (int kk = 0; kk < NKK; ++kk) { s[kk] = __expf(s[kk] - m); l += s[kk]; }
        float inv = 1.0f / l;
        float o[HD];
#pragma unroll
        for (int d = 0; d < HD; ++d) o[d] = 0.0f;
#pragma unroll
        for (int kk = 0; kk < NKK; ++kk) {
            float p = s[kk];
#pragma unroll
            for (int d = 0; d < HD; ++d) o[d] += p * sV[kk * DD + h * HD + d];
        }
        float* orow = out + ((size_t)grp * NKK + qi) * DD + h * HD;
#pragma unroll
        for (int d = 0; d < HD; ++d) orow[d] = o[d] * inv;
    }
}

// =====================================================================
// Across-attention: 68 sequences of length NQ=300 (stride NKK in token idx)
// One block per (b, nk, h, qchunk of 128); K/V staged in LDS; online softmax.
// =====================================================================
#define ACQ 128
__launch_bounds__(ACQ)
__global__ void attn_across_kernel(const float* __restrict__ qkv, float* __restrict__ out) {
    __shared__ float sK[NQ * HD], sV[NQ * HD];
    int bid = blockIdx.x;
    int qc  = bid % 3;
    int h   = (bid / 3) % NH;
    int nk  = (bid / (3 * NH)) % NKK;
    int b   = bid / (3 * NH * NKK);
    for (int i = threadIdx.x; i < NQ * HD; i += blockDim.x) {
        int j = i / HD, d = i % HD;
        size_t t = (size_t)(b * NQ + j) * NKK + nk;
        const float* row = qkv + t * (3 * DD);
        sK[i] = row[DD + h * HD + d];
        sV[i] = row[2 * DD + h * HD + d];
    }
    __syncthreads();
    int qi = qc * ACQ + threadIdx.x;
    if (qi < NQ) {
        size_t tq = (size_t)(b * NQ + qi) * NKK + nk;
        const float* qrow = qkv + tq * (3 * DD) + h * HD;
        float q[HD];
#pragma unroll
        for (int d = 0; d < HD; ++d) q[d] = qrow[d] * 0.17677669529663687f;
        float m = -1e30f, l = 0.0f;
        float o[HD];
#pragma unroll
        for (int d = 0; d < HD; ++d) o[d] = 0.0f;
        for (int j = 0; j < NQ; ++j) {
            float s = 0.0f;
#pragma unroll
            for (int d = 0; d < HD; ++d) s += q[d] * sK[j * HD + d];
            float mn   = fmaxf(m, s);
            float corr = __expf(m - mn);
            float p    = __expf(s - mn);
            l = l * corr + p;
#pragma unroll
            for (int d = 0; d < HD; ++d) o[d] = o[d] * corr + p * sV[j * HD + d];
            m = mn;
        }
        float inv = 1.0f / l;
        float* orow = out + tq * DD + h * HD;
#pragma unroll
        for (int d = 0; d < HD; ++d) orow[d] = o[d] * inv;
    }
}

// =====================================================================
// MSDeform sampling: one wave per (token, head); lane = channel.
// val: [b][lv][h*32+c]; off: [tok][256]; aw: [tok][128]; ref: [b][lq][l][2]
// =====================================================================
__launch_bounds__(256)
__global__ void msda_sample_kernel(const float* __restrict__ val,
                                   const float* __restrict__ off,
                                   const float* __restrict__ aw,
                                   const float* __restrict__ ref,
                                   float* __restrict__ out) {
    const int Hs[NLV]     = {92, 46, 23, 12};
    const int Wsz[NLV]    = {92, 46, 23, 12};
    const int starts[NLV] = {0, 8464, 10580, 11109};
    int t    = blockIdx.x;
    int h    = threadIdx.x >> 5;
    int lane = threadIdx.x & 31;
    int b  = t / LQ;
    int lq = t - b * LQ;

    const float* awp = aw + (size_t)t * (NH * NLV * NP) + h * (NLV * NP);
    float wmax = -1e30f;
#pragma unroll
    for (int i = 0; i < NLV * NP; ++i) wmax = fmaxf(wmax, awp[i]);
    float w[NLV * NP];
    float wsum = 0.0f;
#pragma unroll
    for (int i = 0; i < NLV * NP; ++i) { w[i] = __expf(awp[i] - wmax); wsum += w[i]; }
    float winv = 1.0f / wsum;

    const float* offp = off + (size_t)t * DD + h * (NLV * NP * 2);
    const float* refp = ref + ((size_t)b * LQ + lq) * (NLV * 2);

    float acc = 0.0f;
#pragma unroll
    for (int l = 0; l < NLV; ++l) {
        int Hl = Hs[l], Wl = Wsz[l];
        const float* vbase = val + ((size_t)b * LV + starts[l]) * DD + h * HD + lane;
        float rx = refp[l * 2 + 0], ry = refp[l * 2 + 1];
#pragma unroll
        for (int p = 0; p < NP; ++p) {
            float lx = rx + offp[l * (NP * 2) + p * 2 + 0] / (float)Wl;
            float ly = ry + offp[l * (NP * 2) + p * 2 + 1] / (float)Hl;
            float x = lx * (float)Wl - 0.5f;
            float y = ly * (float)Hl - 0.5f;
            float x0 = floorf(x), y0 = floorf(y);
            float wx = x - x0, wy = y - y0;
            int x0i = (int)x0, y0i = (int)y0;
            float v00 = 0.0f, v01 = 0.0f, v10 = 0.0f, v11 = 0.0f;
            if (x0i >= 0 && x0i < Wl && y0i >= 0 && y0i < Hl)
                v00 = vbase[(size_t)(y0i * Wl + x0i) * DD];
            if (x0i + 1 >= 0 && x0i + 1 < Wl && y0i >= 0 && y0i < Hl)
                v01 = vbase[(size_t)(y0i * Wl + x0i + 1) * DD];
            if (x0i >= 0 && x0i < Wl && y0i + 1 >= 0 && y0i + 1 < Hl)
                v10 = vbase[(size_t)((y0i + 1) * Wl + x0i) * DD];
            if (x0i + 1 >= 0 && x0i + 1 < Wl && y0i + 1 >= 0 && y0i + 1 < Hl)
                v11 = vbase[(size_t)((y0i + 1) * Wl + x0i + 1) * DD];
            float s = v00 * (1.0f - wx) * (1.0f - wy) + v01 * wx * (1.0f - wy)
                    + v10 * (1.0f - wx) * wy          + v11 * wx * wy;
            acc += w[l * NP + p] * winv * s;
        }
    }
    out[(size_t)t * DD + h * HD + lane] = acc;
}

// =====================================================================
// Residual + LayerNorm: out = LN(a + b) * g + beta ; one wave per token
// =====================================================================
__launch_bounds__(256)
__global__ void residual_ln_kernel(const float* __restrict__ a, const float* __restrict__ bres,
                                   const float* __restrict__ g, const float* __restrict__ beta,
                                   float* __restrict__ out, int ntok) {
    int t = blockIdx.x * 8 + (threadIdx.x >> 5);
    if (t >= ntok) return;
    int lane = threadIdx.x & 31;
    const float* ap = a    + (size_t)t * DD + lane * 8;
    const float* bp = bres + (size_t)t * DD + lane * 8;
    float v[8], s = 0.0f;
#pragma unroll
    for (int i = 0; i < 8; ++i) { v[i] = ap[i] + bp[i]; s += v[i]; }
#pragma unroll
    for (int o = 16; o > 0; o >>= 1) s += __shfl_xor(s, o, 32);
    float mean = s * (1.0f / 256.0f);
    float qv = 0.0f;
#pragma unroll
    for (int i = 0; i < 8; ++i) { float d = v[i] - mean; qv += d * d; }
#pragma unroll
    for (int o = 16; o > 0; o >>= 1) qv += __shfl_xor(qv, o, 32);
    float rstd = rsqrtf(qv * (1.0f / 256.0f) + 1e-5f);
    float* op = out + (size_t)t * DD + lane * 8;
#pragma unroll
    for (int i = 0; i < 8; ++i) {
        int d = lane * 8 + i;
        op[i] = (v[i] - mean) * rstd * g[d] + beta[d];
    }
}

// =====================================================================
// Gate combine + LayerNorm: out = LN(g1*x2 + g2*t2) * g + beta
// gates: [tok][512]  (g1 = [:256], g2 = [256:])
// =====================================================================
__launch_bounds__(256)
__global__ void gate_ln_kernel(const float* __restrict__ gates,
                               const float* __restrict__ x2, const float* __restrict__ t2,
                               const float* __restrict__ g, const float* __restrict__ beta,
                               float* __restrict__ out, int ntok) {
    int t = blockIdx.x * 8 + (threadIdx.x >> 5);
    if (t >= ntok) return;
    int lane = threadIdx.x & 31;
    const float* gp = gates + (size_t)t * (2 * DD) + lane * 8;
    const float* xp = x2    + (size_t)t * DD + lane * 8;
    const float* tp = t2    + (size_t)t * DD + lane * 8;
    float v[8], s = 0.0f;
#pragma unroll
    for (int i = 0; i < 8; ++i) {
        v[i] = gp[i] * xp[i] + gp[DD + i] * tp[i];
        s += v[i];
    }
#pragma unroll
    for (int o = 16; o > 0; o >>= 1) s += __shfl_xor(s, o, 32);
    float mean = s * (1.0f / 256.0f);
    float qv = 0.0f;
#pragma unroll
    for (int i = 0; i < 8; ++i) { float d = v[i] - mean; qv += d * d; }
#pragma unroll
    for (int o = 16; o > 0; o >>= 1) qv += __shfl_xor(qv, o, 32);
    float rstd = rsqrtf(qv * (1.0f / 256.0f) + 1e-5f);
    float* op = out + (size_t)t * DD + lane * 8;
#pragma unroll
    for (int i = 0; i < 8; ++i) {
        int d = lane * 8 + i;
        op[i] = (v[i] - mean) * rstd * g[d] + beta[d];
    }
}

// =====================================================================
// host-side launch helper (K dispatch: 256 or 1024)
// =====================================================================
static inline void launch_gemm(const float* A, int lda, const float* W, int ldw,
                               const float* bias, const float* Cin, float* C, int ldc,
                               int M, int N, int K, int act, hipStream_t s) {
    int mt = (M + 15) / 16, nt = N / 64;
    int blocks = (mt * nt + 7) / 8;
    if (K == 1024)
        gemm_wmma_kernel<1024><<<blocks, 256, 0, s>>>(A, lda, W, ldw, bias, Cin, C, ldc, M, N, act);
    else
        gemm_wmma_kernel<256><<<blocks, 256, 0, s>>>(A, lda, W, ldw, bias, Cin, C, ldc, M, N, act);
}

extern "C" void kernel_launch(void* const* d_in, const int* in_sizes, int n_in,
                              void* d_out, int out_size, void* d_ws, size_t ws_size,
                              hipStream_t stream) {
    (void)in_sizes; (void)n_in; (void)out_size; (void)ws_size;
    const float* tgt    = (const float*)d_in[0];
    const float* pos    = (const float*)d_in[1];
    const float* ref    = (const float*)d_in[2];
    const float* memry  = (const float*)d_in[3];
    // d_in[4] = spatial shapes (int), hardcoded
    const float* wa_in_w  = (const float*)d_in[5];
    const float* wa_in_b  = (const float*)d_in[6];
    const float* wa_out_w = (const float*)d_in[7];
    const float* wa_out_b = (const float*)d_in[8];
    const float* ac_in_w  = (const float*)d_in[9];
    const float* ac_in_b  = (const float*)d_in[10];
    const float* ac_out_w = (const float*)d_in[11];
    const float* ac_out_b = (const float*)d_in[12];
    const float* ms_val_w = (const float*)d_in[13];
    const float* ms_val_b = (const float*)d_in[14];
    const float* ms_off_w = (const float*)d_in[15];
    const float* ms_off_b = (const float*)d_in[16];
    const float* ms_aw_w  = (const float*)d_in[17];
    const float* ms_aw_b  = (const float*)d_in[18];
    const float* ms_out_w = (const float*)d_in[19];
    const float* ms_out_b = (const float*)d_in[20];
    const float* gate_w   = (const float*)d_in[21];
    const float* gate_b   = (const float*)d_in[22];
    const float* l1_w     = (const float*)d_in[23];
    const float* l1_b     = (const float*)d_in[24];
    const float* l2_w     = (const float*)d_in[25];
    const float* l2_b     = (const float*)d_in[26];
    const float* gn_g     = (const float*)d_in[27];
    const float* wn_g     = (const float*)d_in[28];
    const float* an_g     = (const float*)d_in[29];
    const float* n2_g     = (const float*)d_in[30];
    const float* gn_b     = (const float*)d_in[31];
    const float* wn_b     = (const float*)d_in[32];
    const float* an_b     = (const float*)d_in[33];
    const float* n2_b     = (const float*)d_in[34];
    float* outp = (float*)d_out;

    // workspace layout (floats):  ~192 MB total
    float* ws = (float*)d_ws;
    float* X  = ws;                                  // 20400*256   state
    float* S1 = X  + (size_t)NTOK * DD;              // 20400*1024  qkv / off|aw|samp / gates / ffn-hidden
    float* S2 = S1 + (size_t)NTOK * FF;              // 45012*256   val / t2-tmp / f
    float* S3 = S2 + (size_t)MVAL * DD;              // 20400*256   attn-out / t2
    float* S4 = S3 + (size_t)NTOK * DD;              // 20400*256   x2 = x + pos
    float* OFF   = S1;
    float* AW    = S1 + (size_t)NTOK * DD;
    float* SAMP  = AW + (size_t)NTOK * (NH * NLV * NP);
    float* GATES = S1;

    int nelem = NTOK * DD;
    int addBlocks = (nelem + 255) / 256;
    int lnBlocks  = (NTOK + 7) / 8;

    // ---- X = tgt + pos -------------------------------------------------
    add_kernel<<<addBlocks, 256, 0, stream>>>(tgt, pos, X, nelem);

    // ---- within attention ---------------------------------------------
    launch_gemm(X, DD, wa_in_w, DD, wa_in_b, nullptr, S1, 3 * DD, NTOK, 3 * DD, DD, 0, stream);
    attn_within_kernel<<<BSZ * NQ, 256, 0, stream>>>(S1, S3);
    launch_gemm(S3, DD, wa_out_w, DD, wa_out_b, nullptr, S2, DD, NTOK, DD, DD, 0, stream);
    residual_ln_kernel<<<lnBlocks, 256, 0, stream>>>(X, S2, wn_g, wn_b, X, NTOK);

    // ---- across attention ----------------------------------------------
    launch_gemm(X, DD, ac_in_w, DD, ac_in_b, nullptr, S1, 3 * DD, NTOK, 3 * DD, DD, 0, stream);
    attn_across_kernel<<<BSZ * NKK * NH * 3, ACQ, 0, stream>>>(S1, S3);
    launch_gemm(S3, DD, ac_out_w, DD, ac_out_b, nullptr, S2, DD, NTOK, DD, DD, 0, stream);
    residual_ln_kernel<<<lnBlocks, 256, 0, stream>>>(X, S2, an_g, an_b, X, NTOK);

    // ---- x2 = x + pos ---------------------------------------------------
    add_kernel<<<addBlocks, 256, 0, stream>>>(X, pos, S4, nelem);

    // ---- MS deformable attention ---------------------------------------
    launch_gemm(memry, DD, ms_val_w, DD, ms_val_b, nullptr, S2, DD, MVAL, DD, DD, 0, stream);
    launch_gemm(S4, DD, ms_off_w, DD, ms_off_b, nullptr, OFF, DD, NTOK, DD, DD, 0, stream);
    launch_gemm(S4, DD, ms_aw_w, DD, ms_aw_b, nullptr, AW, NH * NLV * NP, NTOK, NH * NLV * NP, DD, 0, stream);
    msda_sample_kernel<<<NTOK, 256, 0, stream>>>(S2, OFF, AW, ref, SAMP);
    launch_gemm(SAMP, DD, ms_out_w, DD, ms_out_b, nullptr, S3, DD, NTOK, DD, DD, 0, stream);

    // ---- gating ---------------------------------------------------------
    launch_gemm(S4, DD, gate_w, 2 * DD, gate_b, nullptr, GATES, 2 * DD, NTOK, 2 * DD, DD, 0, stream);
    launch_gemm(S3, DD, gate_w + DD, 2 * DD, nullptr, GATES, GATES, 2 * DD, NTOK, 2 * DD, DD, 2, stream);
    gate_ln_kernel<<<lnBlocks, 256, 0, stream>>>(GATES, S4, S3, gn_g, gn_b, X, NTOK);

    // ---- FFN ------------------------------------------------------------
    launch_gemm(X, DD, l1_w, DD, l1_b, nullptr, S1, FF, NTOK, FF, DD, 1, stream);
    launch_gemm(S1, FF, l2_w, FF, l2_b, nullptr, S2, DD, NTOK, DD, FF, 0, stream);
    residual_ln_kernel<<<lnBlocks, 256, 0, stream>>>(X, S2, n2_g, n2_b, outp, NTOK);
}
